// DecoderLayer_43654047597074
// MI455X (gfx1250) — compile-verified, matrix operand order checked by reading, not run on hardware
//
#include <hip/hip_runtime.h>

typedef unsigned short u16;
typedef __attribute__((ext_vector_type(16))) __bf16 v16bf;
typedef __attribute__((ext_vector_type(8)))  float  v8f;

// ---------------------------------------------------------------- helpers

__device__ __forceinline__ u16 f2bf(float x) {
    unsigned int u = __float_as_uint(x);
    u = (u + 0x7FFFu + ((u >> 16) & 1u)) >> 16;
    return (u16)u;
}

__device__ __forceinline__ v8f wmma_bf16(v16bf a, v16bf b, v8f c) {
    // D = A(16x32 bf16) * B(32x16 bf16) + C(16x16 f32)
    return __builtin_amdgcn_wmma_f32_16x16x32_bf16(
        /*neg_a=*/false, a, /*neg_b=*/false, b,
        /*c_mod=*/(short)0, c, /*reuse_a=*/false, /*reuse_b=*/false);
}

// Load one 16x32 bf16 operand fragment from LDS (row-major-in-K, pitch `stride`).
// Lane l (<16): row rowBase+l, K = {0..7, 16..23}+kbase ; lane l>=16: K shifted by 8.
__device__ __forceinline__ v16bf load_frag(const u16* base, int rowBase, int stride, int kbase) {
    int lane = threadIdx.x & 31;
    const u16* p = base + (size_t)(rowBase + (lane & 15)) * stride + kbase + ((lane >> 4) * 8);
    union { v16bf v; unsigned int d[8]; } u;
    const unsigned int* q0 = (const unsigned int*)p;
    const unsigned int* q1 = (const unsigned int*)(p + 16);
#pragma unroll
    for (int i = 0; i < 4; ++i) u.d[i] = q0[i];
#pragma unroll
    for (int i = 0; i < 4; ++i) u.d[4 + i] = q1[i];
    return u.v;
}

// ---------------------------------------------------------------- cvt kernels

__global__ void cvt_bf16(const float* __restrict__ in, u16* __restrict__ out, int n) {
    int i = blockIdx.x * blockDim.x + threadIdx.x;
    if (i < n) out[i] = f2bf(in[i]);
}

// W[K][N] f32 -> Wt[N][K] bf16 (32x32 tiles, 256 threads)
__global__ void transpose_cvt(const float* __restrict__ W, u16* __restrict__ Wt, int K, int N) {
    __shared__ float tile[32][33];
    int n0 = blockIdx.x * 32, k0 = blockIdx.y * 32;
    int c = threadIdx.x & 31, r0 = threadIdx.x >> 5;
#pragma unroll
    for (int i = 0; i < 4; ++i)
        tile[r0 + 8 * i][c] = W[(size_t)(k0 + r0 + 8 * i) * N + n0 + c];
    __syncthreads();
#pragma unroll
    for (int i = 0; i < 4; ++i)
        Wt[(size_t)(n0 + r0 + 8 * i) * K + k0 + c] = f2bf(tile[c][r0 + 8 * i]);
}

// ---------------------------------------------------------------- WMMA GEMM
// out[M,N] = act( A[M,K]bf16 @ Wt[N,K]bf16^T + bias[N] ), block tile 128x128, BK=64.
// 8 waves: 4 (M) x 2 (N); wave tile 32x64 = 2x4 WMMA tiles, 16 WMMA per stage.
// Software pipelined: next tile's global loads issued before WMMA block,
// committed to LDS after the read barrier (register double buffering).

template<bool RELU, bool WF, bool WB>
__global__ __launch_bounds__(256) void gemm_bf16(
    const u16* __restrict__ A, const u16* __restrict__ Wt,
    const float* __restrict__ bias,
    float* __restrict__ outF, u16* __restrict__ outB,
    int Mdim, int Ndim, int Kdim)
{
    __shared__ u16 As[128][72];
    __shared__ u16 Bs[128][72];

    int n0 = blockIdx.x * 128, m0 = blockIdx.y * 128;
    int t = threadIdx.x, wave = t >> 5, lane = t & 31;
    int ln_ = lane & 15, lh = lane >> 4;
    int wm = wave & 3, wn = wave >> 2;

    v8f zero = {};
    v8f acc[2][4];
#pragma unroll
    for (int i = 0; i < 2; ++i)
#pragma unroll
        for (int j = 0; j < 4; ++j) acc[i][j] = zero;

    int arow = t >> 1, acb = (t & 1) * 32;   // each thread: 32 u16 of A and of B
    const u16* gA = A  + (size_t)(m0 + arow) * Kdim + acb;
    const u16* gB = Wt + (size_t)(n0 + arow) * Kdim + acb;

    uint4 ar[4], br[4];
    {   // prologue: stage tile k0=0
        const uint4* sA = (const uint4*)gA;
        const uint4* sB = (const uint4*)gB;
#pragma unroll
        for (int i = 0; i < 4; ++i) { ar[i] = sA[i]; br[i] = sB[i]; }
        uint4* dA = (uint4*)&As[arow][acb];
        uint4* dB = (uint4*)&Bs[arow][acb];
#pragma unroll
        for (int i = 0; i < 4; ++i) { dA[i] = ar[i]; dB[i] = br[i]; }
    }

    int ksteps = Kdim >> 6;
    for (int kt = 0; kt < ksteps; ++kt) {
        __syncthreads();   // staged tile visible
        bool has = (kt + 1) < ksteps;
        if (has) {         // prefetch next tile into registers (overlaps with WMMAs)
            const uint4* sA = (const uint4*)(gA + (size_t)(kt + 1) * 64);
            const uint4* sB = (const uint4*)(gB + (size_t)(kt + 1) * 64);
#pragma unroll
            for (int i = 0; i < 4; ++i) { ar[i] = sA[i]; br[i] = sB[i]; }
        }
#pragma unroll
        for (int kc = 0; kc < 2; ++kc) {
            v16bf af[2], bfr[4];
#pragma unroll
            for (int i = 0; i < 2; ++i) af[i] = load_frag(&As[0][0], wm * 32 + i * 16, 72, kc * 32);
#pragma unroll
            for (int j = 0; j < 4; ++j) bfr[j] = load_frag(&Bs[0][0], wn * 64 + j * 16, 72, kc * 32);
#pragma unroll
            for (int i = 0; i < 2; ++i)
#pragma unroll
                for (int j = 0; j < 4; ++j)
                    acc[i][j] = wmma_bf16(af[i], bfr[j], acc[i][j]);
        }
        __syncthreads();   // everyone done reading LDS
        if (has) {
            uint4* dA = (uint4*)&As[arow][acb];
            uint4* dB = (uint4*)&Bs[arow][acb];
#pragma unroll
            for (int i = 0; i < 4; ++i) { dA[i] = ar[i]; dB[i] = br[i]; }
        }
    }

    // epilogue: C layout -> lane ln_ = col, VGPR r = row (+8 for upper lane half)
#pragma unroll
    for (int i = 0; i < 2; ++i) {
#pragma unroll
        for (int j = 0; j < 4; ++j) {
            int ncol = n0 + wn * 64 + j * 16 + ln_;
            float bv = bias[ncol];
            size_t idx = (size_t)(m0 + wm * 32 + i * 16 + lh * 8) * Ndim + ncol;
#pragma unroll
            for (int r = 0; r < 8; ++r) {
                float v = acc[i][j][r] + bv;
                if (RELU) v = fmaxf(v, 0.0f);
                if (WF) outF[idx] = v;
                if (WB) outB[idx] = f2bf(v);
                idx += Ndim;
            }
        }
    }
}

// ---------------------------------------------------------------- flash attention
// HD=64 fixed. Block = 256 thr = 8 waves; block handles 128 q-rows for one (b,h).
// Each wave owns 16 q-rows (full 64-wide k tiles). Online softmax in registers.
// K/V tile j+1 prefetched to registers before compute of tile j.

__global__ __launch_bounds__(256) void flash_attn(
    const u16* __restrict__ qb, int q_bs, int q_hs, int q_ss,
    const u16* __restrict__ kb, int k_bs, int k_hs, int k_ss,
    const u16* __restrict__ vb, int v_bs, int v_hs, int v_ss,
    const float* __restrict__ mask,   // [S,S] additive
    u16* __restrict__ ctx,            // [B,S,Dtot] bf16, head h at col h*64
    int S, int Dtot)
{
    int qt = blockIdx.x, h = blockIdx.y, b = blockIdx.z;
    int q0 = qt * 128;
    int t = threadIdx.x, wave = t >> 5, lane = t & 31;
    int ln_ = lane & 15, lh = lane >> 4;

    __shared__ u16 Qs[128][72];
    __shared__ u16 Ks[64][72];        // [k-idx][d]
    __shared__ u16 Vt[64][72];        // [d][k-idx]  (transposed V)
    __shared__ u16 Ps[8][16][72];     // per-wave P tile [q-row][k-idx]

    const u16* qbase = qb + (size_t)b * q_bs + (size_t)h * q_hs;
    const u16* kbase = kb + (size_t)b * k_bs + (size_t)h * k_hs;
    const u16* vbase = vb + (size_t)b * v_bs + (size_t)h * v_hs;

    int srow = t >> 2, scb = (t & 3) * 16;

    {   // stage Q 128x64
        int row = t >> 1, cb = (t & 1) * 32;
        const uint4* s = (const uint4*)(qbase + (size_t)(q0 + row) * q_ss + cb);
        uint4* d = (uint4*)&Qs[row][cb];
        d[0] = s[0]; d[1] = s[1]; d[2] = s[2]; d[3] = s[3];
    }
    {   // stage K/V tile 0
        const uint4* s = (const uint4*)(kbase + (size_t)srow * k_ss + scb);
        uint4* d = (uint4*)&Ks[srow][scb];
        d[0] = s[0]; d[1] = s[1];
        s = (const uint4*)(vbase + (size_t)srow * v_ss + scb);
        uint4 a0 = s[0], a1 = s[1];
        const u16* e0 = (const u16*)&a0;
        const u16* e1 = (const u16*)&a1;
#pragma unroll
        for (int i = 0; i < 8; ++i) Vt[scb + i][srow] = e0[i];
#pragma unroll
        for (int i = 0; i < 8; ++i) Vt[scb + 8 + i][srow] = e1[i];
    }
    __syncthreads();
    v16bf qf0 = load_frag(&Qs[0][0], wave * 16, 72, 0);
    v16bf qf1 = load_frag(&Qs[0][0], wave * 16, 72, 32);

    v8f zero = {};
    v8f o[4];
#pragma unroll
    for (int dt = 0; dt < 4; ++dt) o[dt] = zero;
    float m_[8], l_[8];
#pragma unroll
    for (int r = 0; r < 8; ++r) { m_[r] = -1e30f; l_[r] = 0.0f; }

    int J = S / 64;
    for (int j = 0; j < J; ++j) {
        bool has = (j + 1) < J;
        uint4 kr0, kr1, vr0, vr1;
        if (has) {   // prefetch next K/V tile into registers
            const uint4* s = (const uint4*)(kbase + (size_t)((j + 1) * 64 + srow) * k_ss + scb);
            kr0 = s[0]; kr1 = s[1];
            s = (const uint4*)(vbase + (size_t)((j + 1) * 64 + srow) * v_ss + scb);
            vr0 = s[0]; vr1 = s[1];
        }

        // scores: 16x64 per wave = 4 WMMA tiles, K=64 over d (2 steps)
        v8f sc[4];
#pragma unroll
        for (int nt = 0; nt < 4; ++nt) {
            v16bf kf0 = load_frag(&Ks[0][0], nt * 16, 72, 0);
            v16bf kf1 = load_frag(&Ks[0][0], nt * 16, 72, 32);
            v8f c = zero;
            c = wmma_bf16(qf0, kf0, c);
            c = wmma_bf16(qf1, kf1, c);
            sc[nt] = c;
        }

        // scale + mask + online softmax (row = lh*8+r within wave's 16 rows)
#pragma unroll
        for (int r = 0; r < 8; ++r) {
            int qrow = q0 + wave * 16 + lh * 8 + r;
            const float* mrow = mask + (size_t)qrow * S + j * 64;
            float vmax = -1e30f;
#pragma unroll
            for (int nt = 0; nt < 4; ++nt) {
                float sv = sc[nt][r] * 0.125f + mrow[nt * 16 + ln_];
                sc[nt][r] = sv;
                vmax = fmaxf(vmax, sv);
            }
#pragma unroll
            for (int off = 1; off < 16; off <<= 1)
                vmax = fmaxf(vmax, __shfl_xor(vmax, off, 32));
            float mnew = fmaxf(m_[r], vmax);
            float alpha = __expf(m_[r] - mnew);
            m_[r] = mnew;
            float psum = 0.0f;
#pragma unroll
            for (int nt = 0; nt < 4; ++nt) {
                float p = __expf(sc[nt][r] - mnew);
                sc[nt][r] = p;
                psum += p;
            }
#pragma unroll
            for (int off = 1; off < 16; off <<= 1)
                psum += __shfl_xor(psum, off, 32);
            l_[r] = l_[r] * alpha + psum;
#pragma unroll
            for (int dt = 0; dt < 4; ++dt) o[dt][r] *= alpha;
            // spill P row to this wave's private LDS tile (C-layout -> A-layout via LDS)
            int prow = lh * 8 + r;
#pragma unroll
            for (int nt = 0; nt < 4; ++nt)
                Ps[wave][prow][nt * 16 + ln_] = f2bf(sc[nt][r]);
        }

        // O += P @ V   (same-wave LDS RAW: DS ops are in-order per wave)
        v16bf pf0 = load_frag(&Ps[wave][0][0], 0, 72, 0);
        v16bf pf1 = load_frag(&Ps[wave][0][0], 0, 72, 32);
#pragma unroll
        for (int dt = 0; dt < 4; ++dt) {
            v16bf vf0 = load_frag(&Vt[0][0], dt * 16, 72, 0);
            v16bf vf1 = load_frag(&Vt[0][0], dt * 16, 72, 32);
            o[dt] = wmma_bf16(pf0, vf0, o[dt]);
            o[dt] = wmma_bf16(pf1, vf1, o[dt]);
        }
        __syncthreads();   // all waves done reading Ks/Vt
        if (has) {         // commit prefetched tile
            uint4* d = (uint4*)&Ks[srow][scb];
            d[0] = kr0; d[1] = kr1;
            const u16* e0 = (const u16*)&vr0;
            const u16* e1 = (const u16*)&vr1;
#pragma unroll
            for (int i = 0; i < 8; ++i) Vt[scb + i][srow] = e0[i];
#pragma unroll
            for (int i = 0; i < 8; ++i) Vt[scb + 8 + i][srow] = e1[i];
        }
        __syncthreads();   // staged tile visible
    }

    // normalize and write bf16 context
#pragma unroll
    for (int r = 0; r < 8; ++r) {
        int qrow = q0 + wave * 16 + lh * 8 + r;
        float invl = 1.0f / l_[r];
        size_t base = ((size_t)b * S + qrow) * (size_t)Dtot + h * 64;
#pragma unroll
        for (int dt = 0; dt < 4; ++dt)
            ctx[base + dt * 16 + ln_] = f2bf(o[dt][r] * invl);
    }
}

// ---------------------------------------------------------------- residual + LayerNorm
// x = a + res ; torch-style: std = sqrt(var(ddof=1)) + eps ; out = g*(x-mean)/std + b
template<bool WB>
__global__ __launch_bounds__(256) void ln_kernel(
    const float* __restrict__ a, const float* __restrict__ res,
    const float* __restrict__ g, const float* __restrict__ bta,
    float* __restrict__ outF, u16* __restrict__ outB)
{
    const int DD = 512;
    int row = blockIdx.x, t = threadIdx.x;
    size_t base = (size_t)row * DD;
    float x0 = a[base + t] + res[base + t];
    float x1 = a[base + t + 256] + res[base + t + 256];
    __shared__ float r1[256], r2[256];
    r1[t] = x0 + x1;
    r2[t] = x0 * x0 + x1 * x1;
    __syncthreads();
    for (int off = 128; off > 0; off >>= 1) {
        if (t < off) { r1[t] += r1[t + off]; r2[t] += r2[t + off]; }
        __syncthreads();
    }
    __shared__ float mean_s, inv_s;
    if (t == 0) {
        float mean = r1[0] / 512.0f;
        float var = (r2[0] - 512.0f * mean * mean) / 511.0f;
        var = fmaxf(var, 0.0f);
        mean_s = mean;
        inv_s = 1.0f / (sqrtf(var) + 1e-5f);
    }
    __syncthreads();
    float mean = mean_s, inv = inv_s;
    float o0 = g[t] * (x0 - mean) * inv + bta[t];
    float o1 = g[t + 256] * (x1 - mean) * inv + bta[t + 256];
    outF[base + t] = o0;
    outF[base + t + 256] = o1;
    if (WB) { outB[base + t] = f2bf(o0); outB[base + t + 256] = f2bf(o1); }
}

// ---------------------------------------------------------------- launch

extern "C" void kernel_launch(void* const* d_in, const int* in_sizes, int n_in,
                              void* d_out, int out_size, void* d_ws, size_t ws_size,
                              hipStream_t stream)
{
    (void)in_sizes; (void)n_in; (void)out_size; (void)ws_size;
    const int Bn = 2, Sn = 2048, Dn = 512, Hn = 8, FHn = 2048;
    const int Mn = Bn * Sn;

    const float* x          = (const float*)d_in[0];
    const float* y          = (const float*)d_in[1];
    const float* mask_self  = (const float*)d_in[2];
    const float* mask_cross = (const float*)d_in[3];
    const float* qkv_w      = (const float*)d_in[4];
    const float* qkv_b      = (const float*)d_in[5];
    const float* self_out_w = (const float*)d_in[6];
    const float* self_out_b = (const float*)d_in[7];
    const float* kv_w       = (const float*)d_in[8];
    const float* kv_b       = (const float*)d_in[9];
    const float* q_w        = (const float*)d_in[10];
    const float* q_b        = (const float*)d_in[11];
    const float* cross_out_w= (const float*)d_in[12];
    const float* cross_out_b= (const float*)d_in[13];
    const float* ffn_w1     = (const float*)d_in[14];
    const float* ffn_b1     = (const float*)d_in[15];
    const float* ffn_w2     = (const float*)d_in[16];
    const float* ffn_b2     = (const float*)d_in[17];
    const float* g1 = (const float*)d_in[18]; const float* b1p = (const float*)d_in[19];
    const float* g2 = (const float*)d_in[20]; const float* b2p = (const float*)d_in[21];
    const float* g3 = (const float*)d_in[22]; const float* b3p = (const float*)d_in[23];

    char* wsc = (char*)d_ws;
    size_t off = 0;
    auto alloc_us = [&](size_t n) -> u16* {
        u16* p = (u16*)(wsc + off); off += (n * 2 + 255) & ~(size_t)255; return p;
    };
    auto alloc_f = [&](size_t n) -> float* {
        float* p = (float*)(wsc + off); off += (n * 4 + 255) & ~(size_t)255; return p;
    };

    u16* y_bf    = alloc_us((size_t)Mn * Dn);
    u16* x_bf    = alloc_us((size_t)Mn * Dn);
    u16* wt_qkv  = alloc_us((size_t)3 * Dn * Dn);
    u16* wt_self = alloc_us((size_t)Dn * Dn);
    u16* wt_kv   = alloc_us((size_t)2 * Dn * Dn);
    u16* wt_q    = alloc_us((size_t)Dn * Dn);
    u16* wt_cr   = alloc_us((size_t)Dn * Dn);
    u16* wt_f1   = alloc_us((size_t)FHn * Dn);
    u16* wt_f2   = alloc_us((size_t)Dn * FHn);
    u16* qkv_bf  = alloc_us((size_t)Mn * 3 * Dn);
    u16* vals_bf = alloc_us((size_t)Mn * Dn);
    u16* y1_bf   = alloc_us((size_t)Mn * Dn);
    u16* kv_bf   = alloc_us((size_t)Mn * 2 * Dn);
    u16* qc_bf   = alloc_us((size_t)Mn * Dn);
    u16* ctx_bf  = alloc_us((size_t)Mn * Dn);
    u16* y2_bf   = alloc_us((size_t)Mn * Dn);
    u16* h_bf    = alloc_us((size_t)Mn * FHn);
    float* attn_f  = alloc_f((size_t)Mn * Dn);
    float* y1_f    = alloc_f((size_t)Mn * Dn);
    float* cross_f = alloc_f((size_t)Mn * Dn);
    float* y2_f    = alloc_f((size_t)Mn * Dn);
    float* ffn_f   = alloc_f((size_t)Mn * Dn);

    int nElem = Mn * Dn;
    cvt_bf16<<<nElem / 256, 256, 0, stream>>>(y, y_bf, nElem);
    cvt_bf16<<<nElem / 256, 256, 0, stream>>>(x, x_bf, nElem);

    transpose_cvt<<<dim3(1536 / 32, Dn / 32), 256, 0, stream>>>(qkv_w, wt_qkv, Dn, 1536);
    transpose_cvt<<<dim3(Dn / 32, Dn / 32),   256, 0, stream>>>(self_out_w, wt_self, Dn, Dn);
    transpose_cvt<<<dim3(1024 / 32, Dn / 32), 256, 0, stream>>>(kv_w, wt_kv, Dn, 1024);
    transpose_cvt<<<dim3(Dn / 32, Dn / 32),   256, 0, stream>>>(q_w, wt_q, Dn, Dn);
    transpose_cvt<<<dim3(Dn / 32, Dn / 32),   256, 0, stream>>>(cross_out_w, wt_cr, Dn, Dn);
    transpose_cvt<<<dim3(FHn / 32, Dn / 32),  256, 0, stream>>>(ffn_w1, wt_f1, Dn, FHn);
    transpose_cvt<<<dim3(Dn / 32, FHn / 32),  256, 0, stream>>>(ffn_w2, wt_f2, FHn, Dn);

    // --- self attention ---
    gemm_bf16<false, false, true><<<dim3(1536 / 128, Mn / 128), 256, 0, stream>>>(
        y_bf, wt_qkv, qkv_b, nullptr, qkv_bf, Mn, 1536, Dn);
    flash_attn<<<dim3(Sn / 128, Hn, Bn), 256, 0, stream>>>(
        qkv_bf,       Sn * 1536, 192, 1536,
        qkv_bf + 64,  Sn * 1536, 192, 1536,
        qkv_bf + 128, Sn * 1536, 192, 1536,
        mask_self, vals_bf, Sn, Dn);
    gemm_bf16<false, true, false><<<dim3(Dn / 128, Mn / 128), 256, 0, stream>>>(
        vals_bf, wt_self, self_out_b, attn_f, nullptr, Mn, Dn, Dn);
    ln_kernel<true><<<Mn, 256, 0, stream>>>(attn_f, y, g1, b1p, y1_f, y1_bf);

    // --- cross attention ---
    gemm_bf16<false, false, true><<<dim3(1024 / 128, Mn / 128), 256, 0, stream>>>(
        x_bf, wt_kv, kv_b, nullptr, kv_bf, Mn, 1024, Dn);
    gemm_bf16<false, false, true><<<dim3(Dn / 128, Mn / 128), 256, 0, stream>>>(
        y1_bf, wt_q, q_b, nullptr, qc_bf, Mn, Dn, Dn);
    flash_attn<<<dim3(Sn / 128, Hn, Bn), 256, 0, stream>>>(
        qc_bf,       Sn * 512,  64,  512,
        kv_bf,       Sn * 1024, 128, 1024,
        kv_bf + 64,  Sn * 1024, 128, 1024,
        mask_cross, ctx_bf, Sn, Dn);
    gemm_bf16<false, true, false><<<dim3(Dn / 128, Mn / 128), 256, 0, stream>>>(
        ctx_bf, wt_cr, cross_out_b, cross_f, nullptr, Mn, Dn, Dn);
    ln_kernel<true><<<Mn, 256, 0, stream>>>(cross_f, y1_f, g2, b2p, y2_f, y2_bf);

    // --- FFN ---
    gemm_bf16<true, false, true><<<dim3(FHn / 128, Mn / 128), 256, 0, stream>>>(
        y2_bf, wt_f1, ffn_b1, nullptr, h_bf, Mn, FHn, Dn);
    gemm_bf16<false, true, false><<<dim3(Dn / 128, Mn / 128), 256, 0, stream>>>(
        h_bf, wt_f2, ffn_b2, ffn_f, nullptr, Mn, Dn, FHn);
    ln_kernel<false><<<Mn, 256, 0, stream>>>(ffn_f, y2_f, g3, b3p, (float*)d_out, nullptr);
}
